// RBM_23819888623677
// MI455X (gfx1250) — matrix operand order repaired and checked by reading, not compiled, not run on hardware
//
#include <hip/hip_runtime.h>
#include <hip/hip_fp16.h>
#include <stdint.h>

typedef __attribute__((ext_vector_type(16))) _Float16 v16h;
typedef __attribute__((ext_vector_type(8)))  _Float16 v8h;
typedef __attribute__((ext_vector_type(4)))  _Float16 v4h;
typedef __attribute__((ext_vector_type(8)))  float    v8f;
typedef __attribute__((ext_vector_type(4)))  float    v4f;

#define B_N 16384
#define C_N 5
#define H_N 256
#define V_N 512
#define K1  (C_N * V_N)   // 2560
#define NUM_ITER 2

#define WMMA_F16(Af, Bf, Cf) __builtin_amdgcn_wmma_f32_16x16x32_f16( \
    false, (Af), false, (Bf), (short)0, (Cf), false, false)

// ---------------- deterministic hash RNG (PCG-style output fn) ---------------
__device__ __forceinline__ float urand(uint32_t idx, uint32_t salt) {
    uint32_t x = idx ^ (salt * 0x9E3779B9u);
    x = x * 747796405u + 2891336453u;
    uint32_t w = ((x >> ((x >> 28) + 4u)) ^ x) * 277803737u;
    w = (w >> 22) ^ w;
    return (float)(w >> 8) * (1.0f / 16777216.0f);  // [0,1)
}

// ---- A-fragment loader: row-major f16 matrix, 16x32 tile per ISA layout -----
// lane 0-15 : M=lane,    halves 0-7 -> K = k0+0..7,   halves 8-15 -> K = k0+16..23
// lane 16-31: M=lane-16, halves 0-7 -> K = k0+8..15,  halves 8-15 -> K = k0+24..31
__device__ __forceinline__ v16h load_a_frag(const _Float16* __restrict__ A,
                                            int lda, int m0, int k0, int lane) {
    const _Float16* p = A + (size_t)(m0 + (lane & 15)) * lda + k0 + ((lane >> 4) << 3);
    union { v16h v; v8h h[2]; } u;
    u.h[0] = *(const v8h*)(p);
    u.h[1] = *(const v8h*)(p + 16);
    return u.v;
}

// ---------------- v0 (f32 one-hot, flat == [b][c*512+v]) -> f16 --------------
__global__ void __launch_bounds__(256)
cvt_v0_kernel(const float* __restrict__ v0, _Float16* __restrict__ Vmat, int n4) {
    int i = blockIdx.x * 256 + threadIdx.x;
    if (i >= n4) return;
    v4f x = *(const v4f*)(v0 + (size_t)i * 4);
    v4h y; y.x = (_Float16)x.x; y.y = (_Float16)x.y; y.z = (_Float16)x.z; y.w = (_Float16)x.w;
    *(v4h*)(Vmat + (size_t)i * 4) = y;
}

// ---- pack W (C,H,V) f32 into WMMA B-fragment-contiguous f16 layouts ---------
// Fragment layout for 16-bit 32x16 B tiles:
//   lanes 0-15 : N = lane,    halves 0-7 -> K = 0..7,  halves 8-15 -> K = 16..23
//   lanes 16-31: N = lane-16, halves 0-7 -> K = 8..15, halves 8-15 -> K = 24..31
// Pack element address: ((tile)*32 + lane)*16 + j, tile = kt*NT + nt.
__global__ void __launch_bounds__(256)
pack_W_kernel(const float* __restrict__ W, _Float16* __restrict__ W1p,
              _Float16* __restrict__ W2p) {
    int i = blockIdx.x * 256 + threadIdx.x;
    if (i >= K1 * H_N) return;
    int j    = i & 15;
    int lane = (i >> 4) & 31;
    int tile = i >> 9;
    int kloc = ((lane >> 4) << 3) + ((j < 8) ? j : (8 + j));
    int ncol = lane & 15;
    // W1: logical B is (K1 x H): W1[k=c*512+v][n=h] = W[c][h][v]
    {
        int nt = tile % (H_N / 16), kt = tile / (H_N / 16);
        int k = kt * 32 + kloc;             // 0..2559
        int n = nt * 16 + ncol;             // hidden 0..255
        int c = k / V_N, v = k % V_N;
        W1p[i] = (_Float16)W[((size_t)c * H_N + n) * V_N + v];
    }
    // W2: logical B is (H x K1): W2[k=h][n=c*512+v] = W[c][h][v]
    {
        int nt = tile % (K1 / 16), kt = tile / (K1 / 16);
        int k = kt * 32 + kloc;             // hidden 0..255
        int n = nt * 16 + ncol;             // 0..2559
        int c = n / V_N, v = n % V_N;
        W2p[i] = (_Float16)W[((size_t)c * H_N + k) * V_N + v];
    }
}

// ============================ sample_h GEMM ==================================
// block (32,4): wave w owns a 32(M) x 64(N) tile; grid.x = B/32 row blocks.
// Manual ping-pong: two named fragment sets, no rotation copies.
struct FragSetH { v16h a0, a1, b0, b1, b2, b3; };

__device__ __forceinline__ void load_set_h(FragSetH& s,
                                           const _Float16* __restrict__ Vmat,
                                           int m0, int kt,
                                           const _Float16* __restrict__ bp0,
                                           int lane) {
    s.a0 = load_a_frag(Vmat, K1, m0,      kt * 32, lane);
    s.a1 = load_a_frag(Vmat, K1, m0 + 16, kt * 32, lane);
    const _Float16* p = bp0 + (size_t)kt * 8192;   // 16 tiles * 512 halves
    s.b0 = *(const v16h*)(p);
    s.b1 = *(const v16h*)(p + 512);
    s.b2 = *(const v16h*)(p + 1024);
    s.b3 = *(const v16h*)(p + 1536);
}

__device__ __forceinline__ void mma_set_h(const FragSetH& s, v8f acc[8]) {
    acc[0] = WMMA_F16(s.a0, s.b0, acc[0]);
    acc[1] = WMMA_F16(s.a0, s.b1, acc[1]);
    acc[2] = WMMA_F16(s.a0, s.b2, acc[2]);
    acc[3] = WMMA_F16(s.a0, s.b3, acc[3]);
    acc[4] = WMMA_F16(s.a1, s.b0, acc[4]);
    acc[5] = WMMA_F16(s.a1, s.b1, acc[5]);
    acc[6] = WMMA_F16(s.a1, s.b2, acc[6]);
    acc[7] = WMMA_F16(s.a1, s.b3, acc[7]);
}

__global__ void __launch_bounds__(128)
gemm_sample_h(const _Float16* __restrict__ Vmat, const _Float16* __restrict__ W1p,
              const float* __restrict__ cbias, _Float16* __restrict__ Hmat,
              float* __restrict__ h_out, uint32_t salt) {
    const int lane = threadIdx.x;
    const int wid  = threadIdx.y;
    const int m0   = blockIdx.x * 32;
    const int ntb  = wid * 4;                      // first of 4 N-tiles (of 16)
    const int KT   = K1 / 32;                      // 80 (even)

    v8f acc[8];
#pragma unroll
    for (int i = 0; i < 8; ++i) acc[i] = v8f{};

    const _Float16* bp0 = W1p + (size_t)ntb * 512 + (size_t)lane * 16;

    FragSetH s0, s1;
    load_set_h(s0, Vmat, m0, 0, bp0, lane);
    for (int kt = 0; kt < KT - 2; kt += 2) {
        __builtin_prefetch(Vmat + (size_t)(m0 + (lane & 15)) * K1 + kt * 32 + 192, 0, 1);
        load_set_h(s1, Vmat, m0, kt + 1, bp0, lane);
        mma_set_h(s0, acc);                        // k-step kt
        load_set_h(s0, Vmat, m0, kt + 2, bp0, lane);
        mma_set_h(s1, acc);                        // k-step kt+1
    }
    load_set_h(s1, Vmat, m0, KT - 1, bp0, lane);
    mma_set_h(s0, acc);                            // k-step KT-2
    mma_set_h(s1, acc);                            // k-step KT-1

    // epilogue: bias + sigmoid + bernoulli, write f16 (+ optional f32 out)
    const int nlane = (ntb << 4) + (lane & 15);
#pragma unroll
    for (int mt = 0; mt < 2; ++mt) {
        const int mbase = m0 + mt * 16 + ((lane >> 4) << 3);
#pragma unroll
        for (int t = 0; t < 4; ++t) {
            int n = nlane + t * 16;
            float cb = cbias[n];
#pragma unroll
            for (int r = 0; r < 8; ++r) {
                int m = mbase + r;
                float x = acc[mt * 4 + t][r] + cb;
                float p = 1.0f / (1.0f + __expf(-x));
                float u = urand((uint32_t)(m * H_N + n), salt);
                float hv = (u < p) ? 1.0f : 0.0f;
                Hmat[(size_t)m * H_N + n] = (_Float16)hv;
                if (h_out) h_out[(size_t)m * H_N + n] = hv;
            }
        }
    }
}

// ============================ sample_v GEMM ==================================
// block (32,4): wave owns 16 batches x 16 v's, ALL 5 classes as 5 WMMA chains,
// so the 5 logits of one (b,v) sit in the same lane/row slot -> in-register
// categorical sampling, no logits round-trip. grid = (B/16, V/64).
struct FragSetV { v16h a; v16h b[C_N]; };

__device__ __forceinline__ void load_set_v(FragSetV& s,
                                           const _Float16* __restrict__ Hmat,
                                           int m0, int kt,
                                           const _Float16* __restrict__ bp0,
                                           int lane) {
    s.a = load_a_frag(Hmat, H_N, m0, kt * 32, lane);
    const _Float16* p = bp0 + (size_t)kt * (160 * 512);
#pragma unroll
    for (int c = 0; c < C_N; ++c)
        s.b[c] = *(const v16h*)(p + (size_t)c * (32 * 512));
}

__device__ __forceinline__ void mma_set_v(const FragSetV& s, v8f acc[C_N]) {
#pragma unroll
    for (int c = 0; c < C_N; ++c)
        acc[c] = WMMA_F16(s.a, s.b[c], acc[c]);
}

__global__ void __launch_bounds__(128)
gemm_sample_v(const _Float16* __restrict__ Hmat, const _Float16* __restrict__ W2p,
              const float* __restrict__ bbias, _Float16* __restrict__ Vmat,
              float* __restrict__ v_out, uint32_t salt) {
    const int lane = threadIdx.x;
    const int wid  = threadIdx.y;
    const int m0   = blockIdx.x * 16;
    const int vt   = blockIdx.y * 4 + wid;          // v-tile index (of 16 v's)
    const int KT   = H_N / 32;                      // 8 (even)

    v8f acc[C_N];
#pragma unroll
    for (int c = 0; c < C_N; ++c) acc[c] = v8f{};

    // B tile for class c, k-step kt: W2p + (kt*160 + c*32 + vt)*512
    const _Float16* bp0 = W2p + ((size_t)vt) * 512 + (size_t)lane * 16;

    FragSetV s0, s1;
    load_set_v(s0, Hmat, m0, 0, bp0, lane);
    for (int kt = 0; kt < KT - 2; kt += 2) {
        load_set_v(s1, Hmat, m0, kt + 1, bp0, lane);
        mma_set_v(s0, acc);
        load_set_v(s0, Hmat, m0, kt + 2, bp0, lane);
        mma_set_v(s1, acc);
    }
    load_set_v(s1, Hmat, m0, KT - 1, bp0, lane);
    mma_set_v(s0, acc);
    mma_set_v(s1, acc);

    // epilogue: categorical over C via inverse-CDF with one hashed uniform
    const int nv    = (vt << 4) + (lane & 15);      // v index 0..511
    const int mbase = m0 + ((lane >> 4) << 3);
    float bb[C_N];
#pragma unroll
    for (int c = 0; c < C_N; ++c) bb[c] = bbias[c * V_N + nv];
#pragma unroll
    for (int r = 0; r < 8; ++r) {
        int m = mbase + r;
        float l[C_N], mx = -1e30f;
#pragma unroll
        for (int c = 0; c < C_N; ++c) { l[c] = acc[c][r] + bb[c]; mx = fmaxf(mx, l[c]); }
        float e[C_N], s = 0.0f;
#pragma unroll
        for (int c = 0; c < C_N; ++c) { e[c] = __expf(l[c] - mx); s += e[c]; }
        float u = urand((uint32_t)(m * V_N + nv), salt) * s;
        int sel = 0; float cum = 0.0f;
#pragma unroll
        for (int c = 0; c < C_N - 1; ++c) { cum += e[c]; if (u >= cum) sel = c + 1; }
        size_t base = (size_t)m * K1 + nv;
#pragma unroll
        for (int c = 0; c < C_N; ++c) {
            float oneh = (c == sel) ? 1.0f : 0.0f;
            Vmat[base + (size_t)c * V_N] = (_Float16)oneh;
            if (v_out) v_out[base + (size_t)c * V_N] = oneh;
        }
    }
}

extern "C" void kernel_launch(void* const* d_in, const int* in_sizes, int n_in,
                              void* d_out, int out_size, void* d_ws, size_t ws_size,
                              hipStream_t stream) {
    (void)in_sizes; (void)n_in; (void)out_size; (void)ws_size;
    const float* v0 = (const float*)d_in[0];   // (B, C, V, 1)
    const float* W  = (const float*)d_in[1];   // (C, H, V)
    const float* bb = (const float*)d_in[2];   // (C, V, 1) -> flat c*512+v
    const float* cb = (const float*)d_in[3];   // (H, 1)

    // d_out = [ v (B*C*V) | h (B*H) | h_given_v0 (B*H) ] as f32
    float* out_v  = (float*)d_out;
    float* out_h  = out_v + (size_t)B_N * K1;
    float* out_h0 = out_h + (size_t)B_N * H_N;

    // workspace: Vmat f16 (B x 2560) | Hmat f16 (B x 256) | W1p | W2p
    _Float16* Vmat = (_Float16*)d_ws;
    _Float16* Hmat = Vmat + (size_t)B_N * K1;
    _Float16* W1p  = Hmat + (size_t)B_N * H_N;
    _Float16* W2p  = W1p + (size_t)K1 * H_N;

    // 1) v0 -> f16 A-matrix (same flat ordering)
    {
        int n4 = (B_N * K1) / 4;
        cvt_v0_kernel<<<(n4 + 255) / 256, 256, 0, stream>>>(v0, Vmat, n4);
    }
    // 2) pack W into fragment layouts
    pack_W_kernel<<<(K1 * H_N + 255) / 256, 256, 0, stream>>>(W, W1p, W2p);

    dim3 blk(32, 4);
    // 3) h_given_v0 = sample_h(v0)
    gemm_sample_h<<<dim3(B_N / 32), blk, 0, stream>>>(Vmat, W1p, cb, Hmat, out_h0, 0xA341u);
    // 4) Gibbs chain
    for (int it = 0; it < NUM_ITER; ++it) {
        bool last = (it == NUM_ITER - 1);
        gemm_sample_v<<<dim3(B_N / 16, V_N / 64), blk, 0, stream>>>(
            Hmat, W2p, bb, Vmat, last ? out_v : nullptr, 0x51C7u + 2u * (uint32_t)it);
        gemm_sample_h<<<dim3(B_N / 32), blk, 0, stream>>>(
            Vmat, W1p, cb, Hmat, last ? out_h : nullptr, 0xB797u + 2u * (uint32_t)it);
    }
}
